// ALiBiMultiHeadAttention_35570919146105
// MI455X (gfx1250) — matrix operand order repaired and checked by reading, not compiled, not run on hardware
//
#include <hip/hip_runtime.h>
#include <hip/hip_bf16.h>

typedef __attribute__((ext_vector_type(8)))  __bf16 v8bf;
typedef __attribute__((ext_vector_type(16))) __bf16 v16bf;
typedef __attribute__((ext_vector_type(8)))  float  v8f;

#define BB 2
#define SS 2048
#define DD 1024
#define HH 16
#define HD 64

// ---------------- WMMA helpers (CDNA5 gfx1250, wave32) ----------------

__device__ __forceinline__ v8f wmma_bf16(v16bf a, v16bf b, v8f c) {
  // D = A(16x32) x B(32x16) + C, fp32 accumulate
  return __builtin_amdgcn_wmma_f32_16x16x32_bf16(
      /*neg_a=*/false, a, /*neg_b=*/false, b,
      /*c_mod=*/(short)0, c, /*reuse_a=*/false, /*reuse_b=*/false);
}

// A fragment: 16x32 bf16 tile, row-major source, 'stride' elems per row.
// ISA layout: lanes 0-15 m=lane, VGPR0-3 k=half*8+[0,8), VGPR4-7 k=16+half*8+[0,8)
__device__ __forceinline__ v16bf load_a16(const __bf16* base, int stride) {
  const int lane = threadIdx.x & 31;
  const __bf16* p = base + (lane & 15) * stride + ((lane >> 4) << 3);
  v8bf lo = *(const v8bf*)p;         // k = half*8 .. +7   -> VGPR 0-3
  v8bf hi = *(const v8bf*)(p + 16);  // k = 16+half*8 .. +7-> VGPR 4-7
  v16bf r;
#pragma unroll
  for (int i = 0; i < 8; ++i) { r[i] = lo[i]; r[i + 8] = hi[i]; }
  return r;
}

// A fragment from fp32 source (normalized P in LDS), converted to bf16.
__device__ __forceinline__ v16bf load_a16_f32(const float* base, int stride) {
  const int lane = threadIdx.x & 31;
  const float* p = base + (lane & 15) * stride + ((lane >> 4) << 3);
  v16bf r;
#pragma unroll
  for (int i = 0; i < 8; ++i) { r[i] = (__bf16)p[i]; r[i + 8] = (__bf16)p[16 + i]; }
  return r;
}

// B fragment: 32x16 bf16 tile stored K-major (i.e. B^T row-major: elem(k,n)=base[n*stride+k]).
// ISA layout: lanes 0-15 n=lane, VGPR0-3 k=half*8+[0,8), VGPR4-7 k=16+half*8+[0,8)
__device__ __forceinline__ v16bf load_b16(const __bf16* base, int stride) {
  const int lane = threadIdx.x & 31;
  const __bf16* p = base + (lane & 15) * stride + ((lane >> 4) << 3);
  v8bf lo = *(const v8bf*)p;        // k = half*8 + [0,8)  -> VGPR 0-3
  v8bf hi = *(const v8bf*)(p + 16); // k = 16+half*8 + [0,8) -> VGPR 4-7
  v16bf r;
#pragma unroll
  for (int i = 0; i < 8; ++i) { r[i] = lo[i]; r[i + 8] = hi[i]; }
  return r;
}

// ---------------- conversion kernels ----------------

__global__ void cvt_x_kernel(const float* __restrict__ x, __bf16* __restrict__ xb, int n) {
  int i = blockIdx.x * blockDim.x + threadIdx.x;
  if (i < n) xb[i] = (__bf16)x[i];
}

// Wt[n][k] = W[k][n]  (W: [1024][3072] fp32 -> Wt: [3072][1024] bf16)
__global__ void cvt_wt_kernel(const float* __restrict__ W, __bf16* __restrict__ Wt) {
  int i = blockIdx.x * blockDim.x + threadIdx.x;  // 3072*1024 total
  int nn = i >> 10, kk = i & 1023;
  Wt[i] = (__bf16)W[kk * 3072 + nn];
}

// ---------------- QKV GEMM: [4096 x 1024] @ [1024 x 3072] ----------------
// Scatters result into per-head Q[s][d], K[s][d] (row-major) and Vt[d][s].

__global__ __launch_bounds__(256) void qkv_gemm_kernel(
    const __bf16* __restrict__ xb, const __bf16* __restrict__ Wt,
    __bf16* __restrict__ Qh, __bf16* __restrict__ Kh, __bf16* __restrict__ Vt) {
  const int lane = threadIdx.x & 31;
  const int w    = threadIdx.x >> 5;                 // 8 waves
  const int rowbase = blockIdx.y * 128 + (w >> 1) * 32;
  const int colbase = blockIdx.x * 128 + (w & 1) * 64;

  v8f acc[2][4] = {};
  for (int kb = 0; kb < 1024; kb += 32) {
    v16bf a0 = load_a16(xb + rowbase * 1024 + kb, 1024);
    v16bf a1 = load_a16(xb + (rowbase + 16) * 1024 + kb, 1024);
#pragma unroll
    for (int ni = 0; ni < 4; ++ni) {
      v16bf bt = load_b16(Wt + (colbase + ni * 16) * 1024 + kb, 1024);
      acc[0][ni] = wmma_bf16(a0, bt, acc[0][ni]);
      acc[1][ni] = wmma_bf16(a1, bt, acc[1][ni]);
    }
  }

  const int n = lane & 15, hlf = lane >> 4;
#pragma unroll
  for (int mi = 0; mi < 2; ++mi) {
#pragma unroll
    for (int ni = 0; ni < 4; ++ni) {
#pragma unroll
      for (int v = 0; v < 8; ++v) {
        int row = rowbase + mi * 16 + hlf * 8 + v;   // C layout: m=(lane/16)*8+v
        int col = colbase + ni * 16 + n;             //           n=lane%16
        __bf16 val = (__bf16)acc[mi][ni][v];
        int b = row >> 11, s = row & 2047;
        int chunk = col >> 10, cc = col & 1023;
        int h = cc >> 6, d = cc & 63;
        int bh = b * HH + h;
        if (chunk == 0)      Kh[(bh * SS + s) * HD + d] = val;   // key
        else if (chunk == 1) Qh[(bh * SS + s) * HD + d] = val;   // query
        else                 Vt[(bh * HD + d) * SS + s] = val;   // value, transposed
      }
    }
  }
}

// ---------------- ALiBi causal attention (two-pass flash, per-wave 16 q rows) ----------------

__global__ __launch_bounds__(256) void attn_kernel(
    const __bf16* __restrict__ Qh, const __bf16* __restrict__ Kh,
    const __bf16* __restrict__ Vt,
    float* __restrict__ outp, float* __restrict__ attnp) {
  __shared__ __align__(16) __bf16 Pst[8][16][32];    // per-wave P staging (C-layout -> A-layout)

  const int lane = threadIdx.x & 31;
  const int w    = threadIdx.x >> 5;
  const int h = blockIdx.y, b = blockIdx.z;
  const int bh = b * HH + h;
  const int q0 = blockIdx.x * 128 + w * 16;          // this wave's query rows [q0, q0+16)
  const int qmax = q0 + 15;
  const float slope = exp2f(-0.5f * (float)(h + 1)); // ALiBi: (256^(1/16))^-(h+1)
  const float invscale = 0.03125f;                   // 1/sqrt(1024)

  const __bf16* Qb = Qh + ((size_t)bh * SS + q0) * HD;
  const __bf16* Kb = Kh + (size_t)bh * SS * HD;
  const __bf16* Vb = Vt + (size_t)bh * HD * SS;

  // Q fragments (reused across all key tiles)
  v16bf aq0 = load_a16(Qb, HD);        // dims 0..31
  v16bf aq1 = load_a16(Qb + 32, HD);   // dims 32..63

  const int n = lane & 15;
  const int hlf = lane >> 4;
  const int rbase = q0 + hlf * 8;      // global q row for element v=0

  // ---- pass 1: online row-max and row-sum over causal key range ----
  float mrow[8], lrow[8];
#pragma unroll
  for (int v = 0; v < 8; ++v) { mrow[v] = -3.0e38f; lrow[v] = 0.0f; }

  const int ntiles = (q0 >> 4) + 1;                  // causal: keys <= qmax
  for (int nt = 0; nt < ntiles; ++nt) {
    const int kb = nt * 16;
    const __bf16* kp = Kb + kb * HD;
    if (nt + 1 < ntiles) __builtin_prefetch(kp + 16 * HD, 0, 3);
    v8f sf = {};
    sf = wmma_bf16(aq0, load_b16(kp, HD), sf);
    sf = wmma_bf16(aq1, load_b16(kp + 32, HD), sf);
    const int kc = kb + n;
#pragma unroll
    for (int v = 0; v < 8; ++v) {
      const int qr = rbase + v;
      float s = sf[v] * invscale + slope * (float)(kc - qr);
      if (kc > qr) s = -3.0e38f;
      float smx = s;
#pragma unroll
      for (int off = 8; off > 0; off >>= 1) smx = fmaxf(smx, __shfl_xor(smx, off, 16));
      const float mnew = fmaxf(mrow[v], smx);
      float e = __expf(s - mnew);
#pragma unroll
      for (int off = 8; off > 0; off >>= 1) e += __shfl_xor(e, off, 16);
      lrow[v] = lrow[v] * __expf(mrow[v] - mnew) + e;
      mrow[v] = mnew;
    }
  }
  float invl[8];
#pragma unroll
  for (int v = 0; v < 8; ++v) invl[v] = 1.0f / lrow[v];

  // ---- pass 2: normalized probs -> attn (full row incl. zeros), P@V -> out ----
  float* attrow = attnp + (size_t)bh * SS * SS;
  v8f accO[4] = {};
  for (int kt = 0; kt < 64; ++kt) {                  // 32-key steps
    const bool anyv = (kt * 32) <= qmax;
#pragma unroll
    for (int hh2 = 0; hh2 < 2; ++hh2) {
      const int ntt = kt * 2 + hh2;
      const int kb = ntt * 16;
      const int kc = kb + n;
      if (kb <= qmax) {
        const __bf16* kp = Kb + kb * HD;
        v8f sf = {};
        sf = wmma_bf16(aq0, load_b16(kp, HD), sf);
        sf = wmma_bf16(aq1, load_b16(kp + 32, HD), sf);
#pragma unroll
        for (int v = 0; v < 8; ++v) {
          const int qr = rbase + v;
          float s = sf[v] * invscale + slope * (float)(kc - qr);
          float p = (kc <= qr) ? __expf(s - mrow[v]) * invl[v] : 0.0f;
          attrow[(size_t)qr * SS + kc] = p;
          Pst[w][hlf * 8 + v][hh2 * 16 + n] = (__bf16)p;
        }
      } else {
#pragma unroll
        for (int v = 0; v < 8; ++v) {
          attrow[(size_t)(rbase + v) * SS + kc] = 0.0f;
          if (anyv) Pst[w][hlf * 8 + v][hh2 * 16 + n] = (__bf16)0.0f;
        }
      }
    }
    if (anyv) {
      v16bf pa = load_a16(&Pst[w][0][0], 32);
#pragma unroll
      for (int ni = 0; ni < 4; ++ni) {
        v16bf bv = load_b16(Vb + (ni * 16) * SS + kt * 32, SS);
        accO[ni] = wmma_bf16(pa, bv, accO[ni]);
      }
    }
  }

  // out[b][s][h*64+d]
#pragma unroll
  for (int ni = 0; ni < 4; ++ni) {
#pragma unroll
    for (int v = 0; v < 8; ++v) {
      const int qr = rbase + v;
      const int dd = ni * 16 + n;
      outp[((size_t)b * SS + qr) * DD + h * HD + dd] = accO[ni][v];
    }
  }
}

// ---------------- host launcher ----------------

extern "C" void kernel_launch(void* const* d_in, const int* in_sizes, int n_in,
                              void* d_out, int out_size, void* d_ws, size_t ws_size,
                              hipStream_t stream) {
  const float* x = (const float*)d_in[0];    // [2,2048,1024]
  const float* W = (const float*)d_in[1];    // [1024,3072]
  float* outp  = (float*)d_out;                          // out: 4,194,304 floats
  float* attnp = (float*)d_out + (size_t)BB * SS * DD;   // attn: 134,217,728 floats

  char* ws = (char*)d_ws;
  __bf16* xb = (__bf16*)ws;                                  // 8 MB  [4096][1024]
  __bf16* Wt = (__bf16*)(ws + 8388608);                      // 6 MB  [3072][1024]
  __bf16* Qh = (__bf16*)(ws + 8388608 + 6291456);            // 8 MB  [bh][s][d]
  __bf16* Kh = (__bf16*)(ws + 8388608 + 6291456 + 8388608);  // 8 MB  [bh][s][d]
  __bf16* Vt = (__bf16*)(ws + 8388608 + 6291456 + 16777216); // 8 MB  [bh][d][s]

  const int nx = BB * SS * DD;          // 4194304
  const int nw = DD * 3 * DD;           // 3145728
  cvt_x_kernel<<<(nx + 255) / 256, 256, 0, stream>>>(x, xb, nx);
  cvt_wt_kernel<<<(nw + 255) / 256, 256, 0, stream>>>(W, Wt);

  dim3 gGemm(3 * DD / 128, BB * SS / 128);   // (24, 32)
  qkv_gemm_kernel<<<gGemm, 256, 0, stream>>>(xb, Wt, Qh, Kh, Vt);

  dim3 gAttn(SS / 128, HH, BB);              // (16, 16, 2)
  attn_kernel<<<gAttn, 256, 0, stream>>>(Qh, Kh, Vt, outp, attnp);
}
